// StochasticFiringRouter_72808285602024
// MI455X (gfx1250) — matrix operand order repaired and checked by reading, not compiled, not run on hardware
//
#include <hip/hip_runtime.h>
#include <hip/hip_bf16.h>
#include <math.h>

// Problem constants (match reference)
#define B_TOK  4096
#define H_DIM  1024
#define HG     512      // H/2
#define E_EXP  8
#define DFF_D  4096
#define THRESH 0.1f

typedef __attribute__((ext_vector_type(16))) __bf16 v16bf;
typedef __attribute__((ext_vector_type(8)))  float  v8f;

// GEMM tiling: 128x128 block tile, K-step 32 (one bf16 WMMA K), 256 threads = 8 wave32
#define BM 128
#define BN 128
#define BK 32
#define ROWB 80                      // LDS row stride bytes: 64B data + 16B pad (16B aligned)
#define TILE_BYTES (128 * ROWB)      // 10240 B per tile
// smem layout: [A buf0][B buf0][A buf1][B buf1]
#define ABUF(b) ((b) * 2 * TILE_BYTES)
#define BBUF(b) ((b) * 2 * TILE_BYTES + TILE_BYTES)

__device__ __forceinline__ float sigmoidf_(float v) { return 1.0f / (1.0f + __expf(-v)); }

// Issue async byte-copy of one 128x32 bf16 tile (row-major source, leading dim ld)
// into LDS: 512 x 16B chunks, 2 per thread. Tracked by ASYNCcnt.
__device__ __forceinline__ void stage_tile_async(const __bf16* __restrict__ src,
                                                 int row0, int k0, int ld,
                                                 char* sm, int bufOff, int tid)
{
  #pragma unroll
  for (int q = 0; q < 2; q++) {
    int c  = tid * 2 + q;
    int r  = c >> 2;        // row 0..127
    int ch = c & 3;         // 16B chunk within row
    const __bf16* g = src + (size_t)(row0 + r) * ld + (k0 + ch * 8);
    unsigned lds = (unsigned)(uintptr_t)(sm + bufOff + r * ROWB + ch * 16);
    asm volatile("global_load_async_to_lds_b128 %0, %1, off"
                 :: "v"(lds), "v"(g) : "memory");
  }
}

__device__ __forceinline__ void wait_async_lds() {
  asm volatile("s_wait_asynccnt 0x0" ::: "memory");
}

// C(MxN) = epilogue( A(MxK,bf16,row-major) @ Bt(NxK,bf16,row-major)^T + bias )
// Both tiles staged by async global->LDS byte copies, double buffered.
// Hot loop per k-step: 4 async issues + 12 ds_load_b128 + 8 v_wmma + wait/barrier.
// Epilogue: optional bias, optional SiLU; writes f32 (Cf), bf16 (Cb), or
//           rowscale-accumulate into Cf (skipping rows with scale==0).
__global__ __launch_bounds__(256) void gemm_bf16_wmma(
    const __bf16* __restrict__ A, const __bf16* __restrict__ Bt,
    const float* __restrict__ bias,
    const float* __restrict__ rowscale, int rs_stride,
    float* __restrict__ Cf, __bf16* __restrict__ Cb,
    int N, int Kd, int do_silu)
{
  __shared__ alignas(16) char smem[4 * TILE_BYTES];   // 40 KB

  const int tid  = threadIdx.x;
  const int lane = tid & 31;
  const int wave = tid >> 5;           // 0..7
  const int bm   = blockIdx.x * BM;
  const int bn   = blockIdx.y * BN;
  const int wm   = (wave >> 2) * 64;   // wave row: 0 or 64
  const int wn   = (wave & 3) * 32;    // wave col: 0,32,64,96
  const int half = lane >> 4;          // lane group 0..15 / 16..31
  const int lm   = lane & 15;

  v8f acc[4][2];
  #pragma unroll
  for (int i = 0; i < 4; i++)
    #pragma unroll
    for (int j = 0; j < 2; j++)
      acc[i][j] = (v8f){0.f, 0.f, 0.f, 0.f, 0.f, 0.f, 0.f, 0.f};

  const int ktiles = Kd / BK;

  // ---- Prologue: stage tile 0 (both operands async) ----
  stage_tile_async(A,  bm, 0, Kd, smem, ABUF(0), tid);
  stage_tile_async(Bt, bn, 0, Kd, smem, BBUF(0), tid);
  wait_async_lds();
  __syncthreads();

  // ---- Main loop: double buffered ----
  for (int kt = 0; kt < ktiles; ++kt) {
    const int cur = kt & 1, nxt = cur ^ 1;
    const bool has_next = (kt + 1 < ktiles);
    const int k1 = (kt + 1) * BK;

    if (has_next) {                                        // ASYNCcnt path
      stage_tile_async(A,  bm, k1, Kd, smem, ABUF(nxt), tid);
      stage_tile_async(Bt, bn, k1, Kd, smem, BBUF(nxt), tid);
    }

    // ---- Compute on current buffer (overlaps in-flight async loads) ----
    const char* aT = smem + ABUF(cur);
    const char* bT = smem + BBUF(cur);
    v16bf af[4], bfr[2];
    #pragma unroll
    for (int i = 0; i < 4; i++) {
      const __bf16* row = (const __bf16*)(aT + (wm + i * 16 + lm) * ROWB);
      int kb = half * 8;                      // lanes 16..31: K=8..15 / 24..31
      #pragma unroll
      for (int e = 0; e < 16; e++)
        af[i][e] = row[kb + e + ((e >= 8) ? 8 : 0)];
    }
    #pragma unroll
    for (int j = 0; j < 2; j++) {
      const __bf16* row = (const __bf16*)(bT + (wn + j * 16 + lm) * ROWB);
      int kb = half * 8;                      // lanes 16..31: K=8..15 / 24..31
      #pragma unroll
      for (int e = 0; e < 16; e++)
        bfr[j][e] = row[kb + e + ((e >= 8) ? 8 : 0)];
    }
    #pragma unroll
    for (int i = 0; i < 4; i++)
      #pragma unroll
      for (int j = 0; j < 2; j++)
        acc[i][j] = __builtin_amdgcn_wmma_f32_16x16x32_bf16(
            false, af[i], false, bfr[j], (short)0, acc[i][j], false, false);

    wait_async_lds();     // my async writes into nxt landed in LDS
    __syncthreads();      // all waves done reading cur / writing nxt
  }

  // ---- Epilogue: C/D layout -> VGPR r holds M = r (lanes 0..15) or 8+r (16..31) ----
  #pragma unroll
  for (int i = 0; i < 4; i++) {
    #pragma unroll
    for (int j = 0; j < 2; j++) {
      #pragma unroll
      for (int r = 0; r < 8; r++) {
        int row = bm + wm + i * 16 + half * 8 + r;
        int col = bn + wn + j * 16 + lm;
        float v = acc[i][j][r];
        if (bias)    v += bias[col];
        if (do_silu) v = v * sigmoidf_(v);
        if (rowscale) {
          float s = rowscale[(size_t)row * rs_stride];
          if (s != 0.0f) Cf[(size_t)row * N + col] += s * v;
        } else if (Cb) {
          Cb[(size_t)row * N + col] = (__bf16)v;
        } else {
          Cf[(size_t)row * N + col] = v;
        }
      }
    }
  }
}

// fp32 -> bf16 elementwise (n multiple of 4)
__global__ void cvt_bf16(const float* __restrict__ in, __bf16* __restrict__ out, int n) {
  int i = (blockIdx.x * 256 + threadIdx.x) * 4;
  if (i >= n) return;
  float4 v = *(const float4*)&in[i];
  out[i + 0] = (__bf16)v.x;
  out[i + 1] = (__bf16)v.y;
  out[i + 2] = (__bf16)v.z;
  out[i + 3] = (__bf16)v.w;
}

// W (K,N) f32 row-major -> Wt (N,K) bf16 row-major, batched over blockIdx.z.
// 32x32 LDS-tiled transpose: coalesced reads and writes.
__global__ __launch_bounds__(256) void transpose_cvt_bf16(
    const float* __restrict__ in, __bf16* __restrict__ out, int K, int N)
{
  __shared__ float tile[32][33];
  const int k0 = blockIdx.x * 32;
  const int n0 = blockIdx.y * 32;
  const int tx = threadIdx.x & 31;
  const int ty = threadIdx.x >> 5;     // 0..7
  const float* inp  = in  + (size_t)blockIdx.z * K * N;
  __bf16*      outp = out + (size_t)blockIdx.z * K * N;
  #pragma unroll
  for (int q = 0; q < 4; q++)
    tile[ty + q * 8][tx] = inp[(size_t)(k0 + ty + q * 8) * N + (n0 + tx)];
  __syncthreads();
  #pragma unroll
  for (int q = 0; q < 4; q++)
    outp[(size_t)(n0 + ty + q * 8) * K + (k0 + tx)] = (__bf16)tile[tx][ty + q * 8];
}

// One wave32 per token row: logits = g1 @ gate_w2 + b2, softmax -> gate_out,
// top-2 with firing threshold -> dense w_full and total weight tw.
__global__ __launch_bounds__(256) void gate_finish(
    const float* __restrict__ g1, const float* __restrict__ w2,
    const float* __restrict__ b2,
    float* __restrict__ gate_out, float* __restrict__ w_full,
    float* __restrict__ tw)
{
  int row  = blockIdx.x * 8 + (threadIdx.x >> 5);
  int lane = threadIdx.x & 31;
  const float* g = g1 + (size_t)row * HG;

  float acc[E_EXP];
  #pragma unroll
  for (int e = 0; e < E_EXP; e++) acc[e] = 0.f;
  for (int k = lane; k < HG; k += 32) {
    float gv = g[k];
    const float* wr = w2 + (size_t)k * E_EXP;
    #pragma unroll
    for (int e = 0; e < E_EXP; e++) acc[e] += gv * wr[e];
  }
  #pragma unroll
  for (int off = 16; off > 0; off >>= 1) {
    #pragma unroll
    for (int e = 0; e < E_EXP; e++) acc[e] += __shfl_xor(acc[e], off, 32);
  }

  if (lane == 0) {
    float p[E_EXP];
    float mx = -1e30f;
    #pragma unroll
    for (int e = 0; e < E_EXP; e++) { p[e] = acc[e] + b2[e]; mx = fmaxf(mx, p[e]); }
    float s = 0.f;
    #pragma unroll
    for (int e = 0; e < E_EXP; e++) { p[e] = __expf(p[e] - mx); s += p[e]; }
    float inv = 1.0f / s;
    #pragma unroll
    for (int e = 0; e < E_EXP; e++) { p[e] *= inv; gate_out[(size_t)row * E_EXP + e] = p[e]; }

    // top-2, first-index tie-break (matches jax.lax.top_k)
    int i1 = 0;
    #pragma unroll
    for (int e = 1; e < E_EXP; e++) if (p[e] > p[i1]) i1 = e;
    int i2 = (i1 == 0) ? 1 : 0;
    #pragma unroll
    for (int e = 0; e < E_EXP; e++) if (e != i1 && p[e] > p[i2]) i2 = e;

    float wf[E_EXP];
    #pragma unroll
    for (int e = 0; e < E_EXP; e++) wf[e] = 0.f;
    float tot = 0.f;
    if (p[i1] > THRESH) { wf[i1] = p[i1]; tot += p[i1]; }
    if (p[i2] > THRESH) { wf[i2] = p[i2]; tot += p[i2]; }
    #pragma unroll
    for (int e = 0; e < E_EXP; e++) w_full[(size_t)row * E_EXP + e] = wf[e];
    tw[row] = tot;
  }
}

__global__ void zero_f32(float* __restrict__ p, int n) {
  int i = blockIdx.x * 256 + threadIdx.x;
  if (i < n) p[i] = 0.f;
}

__global__ void finalize_out(const float* __restrict__ x,
                             const float* __restrict__ combined,
                             const float* __restrict__ tw,
                             const float* __restrict__ blend,
                             float* __restrict__ out, int n)
{
  int i = blockIdx.x * 256 + threadIdx.x;
  if (i >= n) return;
  float alpha = 1.0f / (1.0f + __expf(-blend[0]));
  int row = i >> 10;                 // / H_DIM
  float t  = tw[row];
  float xv = x[i];
  float c  = (t > 0.f) ? (combined[i] / t) : xv;   // unfired -> x, blends back to x
  out[i] = alpha * c + (1.0f - alpha) * xv;
}

extern "C" void kernel_launch(void* const* d_in, const int* in_sizes, int n_in,
                              void* d_out, int out_size, void* d_ws, size_t ws_size,
                              hipStream_t stream)
{
  const float* x     = (const float*)d_in[0];
  const float* gw1   = (const float*)d_in[1];
  const float* gb1   = (const float*)d_in[2];
  const float* gw2   = (const float*)d_in[3];
  const float* gb2   = (const float*)d_in[4];
  const float* ew1   = (const float*)d_in[5];
  const float* eb1   = (const float*)d_in[6];
  const float* ew2   = (const float*)d_in[7];
  const float* eb2   = (const float*)d_in[8];
  const float* pw    = (const float*)d_in[9];
  const float* blend = (const float*)d_in[10];

  float* out_main = (float*)d_out;                       // (B, H)
  float* out_gate = out_main + (size_t)B_TOK * H_DIM;    // (B, E)

  // Workspace (byte-based, 256B-aligned slabs), ~217 MB total
  char* wsb = (char*)d_ws;
  size_t o = 0;
  auto alloc = [&](size_t bytes) { char* p = wsb + o; o += (bytes + 255) & ~(size_t)255; return p; };
  __bf16* xb      = (__bf16*)alloc((size_t)B_TOK * H_DIM * 2);          // x as bf16
  __bf16* gw1t    = (__bf16*)alloc((size_t)HG * H_DIM * 2);             // (HG, H)
  __bf16* ew1t    = (__bf16*)alloc((size_t)E_EXP * DFF_D * H_DIM * 2);  // (E, DFF, H)
  __bf16* ew2t    = (__bf16*)alloc((size_t)E_EXP * H_DIM * DFF_D * 2);  // (E, H, DFF)
  __bf16* pwt     = (__bf16*)alloc((size_t)E_EXP * H_DIM * H_DIM * 2);  // (E, H, H)
  float*  g1      = (float*) alloc((size_t)B_TOK * HG * 4);             // gate hidden (f32)
  __bf16* h1b     = (__bf16*)alloc((size_t)B_TOK * DFF_D * 2);          // expert hidden (bf16)
  __bf16* h2b     = (__bf16*)alloc((size_t)B_TOK * H_DIM * 2);          // expert out (bf16)
  float*  combined= (float*) alloc((size_t)B_TOK * H_DIM * 4);
  float*  w_full  = (float*) alloc((size_t)B_TOK * E_EXP * 4);
  float*  tw      = (float*) alloc((size_t)B_TOK * 4);

  dim3 blk(256);

  // 0) one-time (per launch) conversions: x -> bf16, weights -> transposed bf16 (N,K)
  cvt_bf16<<<dim3((B_TOK * H_DIM) / 1024), blk, 0, stream>>>(x, xb, B_TOK * H_DIM);
  transpose_cvt_bf16<<<dim3(H_DIM / 32, HG / 32, 1),      blk, 0, stream>>>(gw1, gw1t, H_DIM, HG);
  transpose_cvt_bf16<<<dim3(H_DIM / 32, DFF_D / 32, E_EXP), blk, 0, stream>>>(ew1, ew1t, H_DIM, DFF_D);
  transpose_cvt_bf16<<<dim3(DFF_D / 32, H_DIM / 32, E_EXP), blk, 0, stream>>>(ew2, ew2t, DFF_D, H_DIM);
  transpose_cvt_bf16<<<dim3(H_DIM / 32, H_DIM / 32, E_EXP), blk, 0, stream>>>(pw, pwt, H_DIM, H_DIM);

  // 1) g1 = silu(x @ gate_w1 + gate_b1)       [4096 x 512, K=1024]
  gemm_bf16_wmma<<<dim3(B_TOK / BM, HG / BN), blk, 0, stream>>>(
      xb, gw1t, gb1, nullptr, 0, g1, nullptr, HG, H_DIM, 1);

  // 2) logits -> softmax -> top-2 routing
  gate_finish<<<dim3(B_TOK / 8), blk, 0, stream>>>(g1, gw2, gb2, out_gate, w_full, tw);

  // 3) combined = 0
  zero_f32<<<dim3((B_TOK * H_DIM) / 256), blk, 0, stream>>>(combined, B_TOK * H_DIM);

  // 4) dense expert pipeline (matches reference loop over all E experts)
  for (int e = 0; e < E_EXP; ++e) {
    gemm_bf16_wmma<<<dim3(B_TOK / BM, DFF_D / BN), blk, 0, stream>>>(
        xb, ew1t + (size_t)e * DFF_D * H_DIM, eb1 + (size_t)e * DFF_D, nullptr, 0,
        nullptr, h1b, DFF_D, H_DIM, 1);
    gemm_bf16_wmma<<<dim3(B_TOK / BM, H_DIM / BN), blk, 0, stream>>>(
        h1b, ew2t + (size_t)e * H_DIM * DFF_D, eb2 + (size_t)e * H_DIM, nullptr, 0,
        nullptr, h2b, H_DIM, DFF_D, 0);
    gemm_bf16_wmma<<<dim3(B_TOK / BM, H_DIM / BN), blk, 0, stream>>>(
        h2b, pwt + (size_t)e * H_DIM * H_DIM, nullptr, w_full + e, E_EXP,
        combined, nullptr, H_DIM, H_DIM, 0);
  }

  // 5) normalize by total routing weight, blend with x
  finalize_out<<<dim3((B_TOK * H_DIM) / 256), blk, 0, stream>>>(
      x, combined, tw, blend, out_main, B_TOK * H_DIM);

  (void)in_sizes; (void)n_in; (void)out_size; (void)ws_size;
}
